// MultiHeadAttention_79053168050784
// MI455X (gfx1250) — compile-verified, matrix operand order checked by reading, not run on hardware
//
#include <hip/hip_runtime.h>
#include <hip/hip_bf16.h>

typedef __bf16 bhalf;
typedef __bf16 v8bf  __attribute__((ext_vector_type(8)));
typedef __bf16 v16bf __attribute__((ext_vector_type(16)));
typedef float  v8f   __attribute__((ext_vector_type(8)));

#define DIM    1024
#define SEQ    2048
#define BATCH  2
#define NHEAD  16
#define HD     64
#define MROWS  (BATCH*SEQ)   // 4096

__device__ __forceinline__ v8f zero8() {
  v8f r;
  #pragma unroll
  for (int e = 0; e < 8; ++e) r[e] = 0.0f;
  return r;
}

__device__ __forceinline__ v16bf bf_join(v8bf lo, v8bf hi) {
  return __builtin_shufflevector(lo, hi, 0,1,2,3,4,5,6,7,8,9,10,11,12,13,14,15);
}

__device__ __forceinline__ v8f wmma_bf16(v16bf a, v16bf b, v8f c) {
  // (neg_a, A, neg_b, B, c_mod, C, reuse_a, reuse_b)
  return __builtin_amdgcn_wmma_f32_16x16x32_bf16(false, a, false, b, (short)0, c, false, false);
}

// A-fragment (16x32 bf16, M x K) loaded from row-major memory.
// Per ISA 7.12.2: lanes 0-15 hold M=lane, K in {0..7, 16..23}; lanes 16-31 hold K in {8..15, 24..31}.
__device__ __forceinline__ v16bf load_a_rm(const bhalf* __restrict__ A, int ldk,
                                           int row, int k0, int lane) {
  const bhalf* p = A + (size_t)row * ldk + k0 + ((lane >> 4) << 3);
  v8bf lo = *(const v8bf*)(p);
  v8bf hi = *(const v8bf*)(p + 16);
  return bf_join(lo, hi);
}

// B-fragment from pre-packed layout: fragment = 32 lanes x 16 contiguous bf16 (512 elems).
__device__ __forceinline__ v16bf load_b_pk(const bhalf* __restrict__ Bp, size_t frag, int lane) {
  const bhalf* p = Bp + frag * 512 + lane * 16;
  v8bf lo = *(const v8bf*)(p);
  v8bf hi = *(const v8bf*)(p + 8);
  return bf_join(lo, hi);
}

// Offset of element (k, n) of a K x N matrix inside the B-fragment-packed buffer.
// KS = K/32. Fragment order: [nt][ks][lane][v].
__device__ __forceinline__ size_t bpack_off(int k, int n, int KS) {
  int nt = n >> 4, ks = k >> 5;
  int koff = k & 31, nn = n & 15;
  int grp = koff >> 3;                       // 0..3
  int lane = nn + ((grp & 1) << 4);          // groups 0,2 -> lanes 0-15; 1,3 -> 16-31
  int v = (koff & 7) + ((grp >> 1) << 3);    // groups 2,3 -> upper 8 values
  return (((size_t)nt * KS + ks) * 32 + lane) * 16 + v;
}

// -------------------- stage 0: conversion / packing --------------------

__global__ __launch_bounds__(256) void conv_x_kernel(const float* __restrict__ x,
                                                     bhalf* __restrict__ xb, int n) {
  int i = blockIdx.x * 256 + threadIdx.x;
  if (i < n) xb[i] = (bhalf)x[i];
}

__global__ __launch_bounds__(256) void pack_w_kernel(const float* __restrict__ W,
                                                     bhalf* __restrict__ Wp, int K, int N) {
  int i = blockIdx.x * 256 + threadIdx.x;
  if (i >= K * N) return;
  int k = i / N, n = i - k * N;
  Wp[bpack_off(k, n, K >> 5)] = (bhalf)W[i];
}

// -------------------- stage 1: fused QKV GEMM --------------------
// C[4096, 3072] = xb[4096,1024] @ [Wq|Wk|Wv]. Wave computes 32x64 (2x4 tiles).
__global__ __launch_bounds__(256) void qkv_gemm(
    const bhalf* __restrict__ X,
    const bhalf* __restrict__ Wqp, const bhalf* __restrict__ Wkp, const bhalf* __restrict__ Wvp,
    bhalf* __restrict__ Qb, bhalf* __restrict__ Kp, bhalf* __restrict__ Vp) {
  const int lane = threadIdx.x & 31, wave = threadIdx.x >> 5;
  const int wm = wave & 1, wn = wave >> 1;
  const int M0 = blockIdx.y * 64 + wm * 32;
  const int N0 = blockIdx.x * 256 + wn * 64;
  const int KS = DIM / 32;

  v8f acc[2][4];
  #pragma unroll
  for (int i = 0; i < 2; ++i)
    #pragma unroll
    for (int j = 0; j < 4; ++j) acc[i][j] = zero8();

  for (int ks = 0; ks < KS; ++ks) {
    v16bf a0 = load_a_rm(X, DIM, M0 + (lane & 15), ks * 32, lane);
    v16bf a1 = load_a_rm(X, DIM, M0 + 16 + (lane & 15), ks * 32, lane);
    #pragma unroll
    for (int j = 0; j < 4; ++j) {
      int ng = N0 + j * 16;
      int mat = ng >> 10;
      int ntl = (ng & 1023) >> 4;
      const bhalf* Bp = (mat == 0) ? Wqp : (mat == 1) ? Wkp : Wvp;
      v16bf bb = load_b_pk(Bp, (size_t)ntl * KS + ks, lane);
      acc[0][j] = wmma_bf16(a0, bb, acc[0][j]);
      acc[1][j] = wmma_bf16(a1, bb, acc[1][j]);
    }
  }

  // Epilogue: route Q -> row-major [B,H,S,64]; K -> Bpack(d,t); V -> Bpack(t,d).
  const int sel = lane >> 4, col = lane & 15;
  #pragma unroll
  for (int i = 0; i < 2; ++i) {
    #pragma unroll
    for (int j = 0; j < 4; ++j) {
      int ng = N0 + j * 16;
      int mat = ng >> 10;
      int c0 = ng & 1023;
      #pragma unroll
      for (int e = 0; e < 8; ++e) {
        int m = M0 + i * 16 + e + sel * 8;
        int b = m >> 11, s = m & (SEQ - 1);
        int c = c0 + col;
        int h = c >> 6, d = c & 63;
        bhalf val = (bhalf)acc[i][j][e];
        size_t bh = (size_t)(b * NHEAD + h);
        if (mat == 0)      Qb[(bh * SEQ + s) * HD + d] = val;
        else if (mat == 1) Kp[bh * (size_t)(SEQ * HD) + bpack_off(d, s, HD / 32)] = val;
        else               Vp[bh * (size_t)(SEQ * HD) + bpack_off(s, d, SEQ / 32)] = val;
      }
    }
  }
}

// -------------------- stage 2: fused attention --------------------
// One WG per (b, h, 32-query block). 128KB LDS score panel (CDNA5 WGP LDS = 320KB).
__global__ __launch_bounds__(256) void attn_kernel(
    const bhalf* __restrict__ Qb, const bhalf* __restrict__ Kp,
    const bhalf* __restrict__ Vp, bhalf* __restrict__ Ob) {
  __shared__ __align__(16) bhalf Psh[32 * SEQ];   // 128 KB score/prob panel
  __shared__ float rowscale[32];

  const int tid = threadIdx.x;
  const int lane = tid & 31, wave = tid >> 5;
  const int bh = blockIdx.x >> 6;
  const int q0 = (blockIdx.x & 63) * 32;
  const int b = bh >> 4, h = bh & 15;
  const bhalf* Q = Qb + (size_t)bh * SEQ * HD;
  const bhalf* K = Kp + (size_t)bh * SEQ * HD;
  const bhalf* V = Vp + (size_t)bh * SEQ * HD;

  // Phase 1: S = Q_blk @ K^T  (waves: 2 M-tiles x 4 key-quarters of 512 keys)
  {
    const int wm = wave & 1;
    const int wk = wave >> 1;
    v16bf aq0 = load_a_rm(Q, HD, q0 + wm * 16 + (lane & 15), 0, lane);
    v16bf aq1 = load_a_rm(Q, HD, q0 + wm * 16 + (lane & 15), 32, lane);
    for (int nt = 0; nt < 32; ++nt) {
      int ntg = wk * 32 + nt;
      v8f acc = zero8();
      v16bf b0 = load_b_pk(K, (size_t)ntg * 2 + 0, lane);
      v16bf b1 = load_b_pk(K, (size_t)ntg * 2 + 1, lane);
      acc = wmma_bf16(aq0, b0, acc);
      acc = wmma_bf16(aq1, b1, acc);
      int key = ntg * 16 + (lane & 15);
      int sel = lane >> 4;
      #pragma unroll
      for (int e = 0; e < 8; ++e) {
        int m = wm * 16 + e + sel * 8;
        Psh[m * SEQ + key] = (bhalf)acc[e];
      }
    }
  }
  __syncthreads();

  // Phase 2: per-row softmax, vectorized b128 LDS traffic; scale = 1/(32*sum)
  {
    const int r = tid >> 3;          // 32 rows, 8 threads per row
    const int ch = tid & 7;          // each thread owns 256 contiguous keys
    bhalf* row = &Psh[r * SEQ + ch * 256];

    float mx = -3.4e38f;
    for (int i = 0; i < 256; i += 8) {
      v8bf v = *(const v8bf*)(row + i);
      #pragma unroll
      for (int e = 0; e < 8; ++e) mx = fmaxf(mx, (float)v[e]);
    }
    #pragma unroll
    for (int o = 1; o < 8; o <<= 1) mx = fmaxf(mx, __shfl_xor(mx, o, 8));

    float sm = 0.0f;
    for (int i = 0; i < 256; i += 8) {
      v8bf v = *(const v8bf*)(row + i);
      v8bf ov;
      #pragma unroll
      for (int e = 0; e < 8; ++e) {
        float ex = __expf((float)v[e] - mx);
        sm += ex;
        ov[e] = (bhalf)ex;
      }
      *(v8bf*)(row + i) = ov;
    }
    #pragma unroll
    for (int o = 1; o < 8; o <<= 1) sm += __shfl_xor(sm, o, 8);
    if (ch == 0) rowscale[r] = 1.0f / (sm * 32.0f);   // /sqrt(DIM_Q) AFTER softmax
  }
  __syncthreads();

  // Phase 3: O = P @ V. 8 waves -> 8 independent 16x16 tiles (2 M x 4 d), full 2048 keys.
  {
    const int ti = wave & 1;        // q-row tile
    const int tj = wave >> 1;       // d tile
    v8f acc = zero8();
    for (int ks = 0; ks < SEQ / 32; ++ks) {
      v16bf a = load_a_rm(Psh, SEQ, ti * 16 + (lane & 15), ks * 32, lane);
      v16bf bb = load_b_pk(V, (size_t)tj * (SEQ / 32) + ks, lane);
      acc = wmma_bf16(a, bb, acc);
    }
    const int sel = lane >> 4, d = tj * 16 + (lane & 15);
    #pragma unroll
    for (int e = 0; e < 8; ++e) {
      int m = ti * 16 + e + sel * 8;
      float val = acc[e] * rowscale[m];
      Ob[((size_t)b * SEQ + q0 + m) * DIM + h * HD + d] = (bhalf)val;
    }
  }
}

// -------------------- stage 3: output projection --------------------
__global__ __launch_bounds__(256) void out_gemm(const bhalf* __restrict__ Ob,
                                                const bhalf* __restrict__ WoP,
                                                float* __restrict__ out) {
  const int lane = threadIdx.x & 31, wave = threadIdx.x >> 5;
  const int wm = wave & 1, wn = wave >> 1;
  const int M0 = blockIdx.y * 64 + wm * 32;
  const int N0 = blockIdx.x * 256 + wn * 64;
  const int KS = DIM / 32;

  v8f acc[2][4];
  #pragma unroll
  for (int i = 0; i < 2; ++i)
    #pragma unroll
    for (int j = 0; j < 4; ++j) acc[i][j] = zero8();

  for (int ks = 0; ks < KS; ++ks) {
    v16bf a0 = load_a_rm(Ob, DIM, M0 + (lane & 15), ks * 32, lane);
    v16bf a1 = load_a_rm(Ob, DIM, M0 + 16 + (lane & 15), ks * 32, lane);
    #pragma unroll
    for (int j = 0; j < 4; ++j) {
      int nt = (N0 >> 4) + j;
      v16bf bb = load_b_pk(WoP, (size_t)nt * KS + ks, lane);
      acc[0][j] = wmma_bf16(a0, bb, acc[0][j]);
      acc[1][j] = wmma_bf16(a1, bb, acc[1][j]);
    }
  }

  const int sel = lane >> 4, col = lane & 15;
  #pragma unroll
  for (int i = 0; i < 2; ++i)
    #pragma unroll
    for (int j = 0; j < 4; ++j)
      #pragma unroll
      for (int e = 0; e < 8; ++e) {
        int m = M0 + i * 16 + e + sel * 8;
        int n = N0 + j * 16 + col;
        out[(size_t)m * DIM + n] = acc[i][j][e];
      }
}

// -------------------- launcher --------------------
extern "C" void kernel_launch(void* const* d_in, const int* in_sizes, int n_in,
                              void* d_out, int out_size, void* d_ws, size_t ws_size,
                              hipStream_t stream) {
  const float* x  = (const float*)d_in[0];
  const float* Wq = (const float*)d_in[1];
  const float* Wk = (const float*)d_in[2];
  const float* Wv = (const float*)d_in[3];
  const float* Wo = (const float*)d_in[4];
  float* out = (float*)d_out;

  char* ws = (char*)d_ws;
  const size_t MB = (size_t)1 << 20;
  bhalf* xb  = (bhalf*)(ws +  0 * MB);   // 8 MB  : x in bf16 row-major
  bhalf* wqp = (bhalf*)(ws +  8 * MB);   // 2 MB  : Wq B-packed
  bhalf* wkp = (bhalf*)(ws + 10 * MB);   // 2 MB  : Wk B-packed
  bhalf* wvp = (bhalf*)(ws + 12 * MB);   // 2 MB  : Wv B-packed
  bhalf* wop = (bhalf*)(ws + 14 * MB);   // 2 MB  : Wo B-packed
  bhalf* Qb  = (bhalf*)(ws + 16 * MB);   // 8 MB  : Q [B,H,S,64] row-major
  bhalf* Kp  = (bhalf*)(ws + 24 * MB);   // 8 MB  : K^T B-packed per (b,h)
  bhalf* Vp  = (bhalf*)(ws + 32 * MB);   // 8 MB  : V B-packed per (b,h)
  bhalf* Ob  = (bhalf*)(ws + 40 * MB);   // 8 MB  : attention out [B,S,1024]

  conv_x_kernel<<<(MROWS * DIM) / 256, 256, 0, stream>>>(x, xb, MROWS * DIM);
  pack_w_kernel<<<(DIM * DIM) / 256, 256, 0, stream>>>(Wq, wqp, DIM, DIM);
  pack_w_kernel<<<(DIM * DIM) / 256, 256, 0, stream>>>(Wk, wkp, DIM, DIM);
  pack_w_kernel<<<(DIM * DIM) / 256, 256, 0, stream>>>(Wv, wvp, DIM, DIM);
  pack_w_kernel<<<(DIM * DIM) / 256, 256, 0, stream>>>(Wo, wop, DIM, DIM);

  qkv_gemm<<<dim3(12, 64), 256, 0, stream>>>(xb, wqp, wkp, wvp, Qb, Kp, Vp);
  attn_kernel<<<BATCH * NHEAD * (SEQ / 32), 256, 0, stream>>>(Qb, Kp, Vp, Ob);
  out_gemm<<<dim3(4, 64), 256, 0, stream>>>(Ob, wop, out);
}
